// SelfAttentionLayer_30408368455794
// MI455X (gfx1250) — matrix operand order repaired and checked by reading, not compiled, not run on hardware
//
#include <hip/hip_runtime.h>
#include <hip/hip_bf16.h>

typedef __attribute__((ext_vector_type(16))) _Float16 v16h;
typedef __attribute__((ext_vector_type(8)))  _Float16 v8h;
typedef __attribute__((ext_vector_type(8)))  float    v8f;

union V16 { v16h v; v8h h8[2]; };

#define N_TOK 16384   // B*H
#define BATCH 2048
#define F     256
#define HS    32
#define KGRP  8                      // split-K groups over the key dimension
#define KEYS_PER_GRP (N_TOK / KGRP)  // 2048

static __device__ __forceinline__ v8f wmma_f16(v16h a, v16h b, v8f c) {
  return __builtin_amdgcn_wmma_f32_16x16x32_f16(false, a, false, b, (short)0, c, false, false);
}

// CDNA5 async global->LDS copy (ASYNCcnt-tracked), 16B per lane.
static __device__ __forceinline__ void async_b128(const _Float16* lds_dst,
                                                  const _Float16* gsrc) {
  unsigned lds_off = (unsigned)(uintptr_t)lds_dst;        // aperture: addr[31:0] = LDS offset
  unsigned long long ga = (unsigned long long)(uintptr_t)gsrc;
  asm volatile("global_load_async_to_lds_b128 %0, %1, off"
               :: "v"(lds_off), "v"(ga) : "memory");
}

// ---------------------------------------------------------------------------
// Kernel 1: f32 -> f16 conversion of x, and transposed f16 copies of Wq/Wk/Wv/Wo
// Wt[p][n*256+k] = W_p[k*256+n]   (so WMMA B-operand columns are contiguous)
// ---------------------------------------------------------------------------
__global__ __launch_bounds__(256) void prep_kernel(
    const float* __restrict__ x,
    const float* __restrict__ Wq, const float* __restrict__ Wk,
    const float* __restrict__ Wv, const float* __restrict__ Wo,
    _Float16* __restrict__ xh, _Float16* __restrict__ Wt) {
  int idx = blockIdx.x * blockDim.x + threadIdx.x;   // 0 .. 524287
  if (idx < BATCH * F) xh[idx] = (_Float16)x[idx];
  if (idx < 4 * F * F) {
    int p = idx >> 16, e = idx & 65535;
    int n = e >> 8, k = e & 255;
    const float* W = (p == 0) ? Wq : (p == 1) ? Wk : (p == 2) ? Wv : Wo;
    Wt[idx] = (_Float16)W[k * F + n];
  }
}

// ---------------------------------------------------------------------------
// Kernel 2: QKV projection (WMMA) + bias + scramble-store.
// One wave per 16x16 output tile; 3 * 128 * 16 = 6144 waves.
// Q,K stored scrambled row-major [16384,32]; V stored scrambled AND transposed
// as Vt[32,16384].
// ---------------------------------------------------------------------------
__global__ __launch_bounds__(256) void proj_kernel(
    const _Float16* __restrict__ xh, const _Float16* __restrict__ Wt,
    const float* __restrict__ bq, const float* __restrict__ bk,
    const float* __restrict__ bv,
    _Float16* __restrict__ Qs, _Float16* __restrict__ Ks,
    _Float16* __restrict__ Vt) {
  int lane = threadIdx.x & 31, wave = threadIdx.x >> 5;
  int wid = blockIdx.x * 8 + wave;            // 0 .. 6143
  int p  = wid / 2048;                        // 0=Q 1=K 2=V
  int t  = wid % 2048;
  int m0 = (t >> 4) * 16, n0 = (t & 15) * 16;
  int h = lane >> 4, ln = lane & 15;
  const _Float16* W = Wt + p * (F * F);

  v8f c = {};
  #pragma unroll
  for (int k0 = 0; k0 < F; k0 += 32) {
    V16 a, b;
    const _Float16* ar = xh + (m0 + ln) * F + k0;         // A: row ln, K=16h..
    a.h8[0] = *(const v8h*)(ar + 16 * h);
    a.h8[1] = *(const v8h*)(ar + 16 * h + 8);
    const _Float16* br = W + (n0 + ln) * F + k0 + 16 * h; // B: col ln, K=16h..
    b.h8[0] = *(const v8h*)(br);
    b.h8[1] = *(const v8h*)(br + 8);
    c = wmma_f16(a.v, b.v, c);
  }

  const float* bias = (p == 0) ? bq : (p == 1) ? bk : bv;
  int col   = n0 + ln;
  float bc  = bias[col];
  int r_off = (col & 31) >> 2;                 // scramble: target row offset
  int c_s   = 8 * (col & 3) + (col >> 5);      // scramble: target column
  #pragma unroll
  for (int i = 0; i < 8; i++) {
    int row = m0 + i + 8 * h;                  // batch row
    float v = c[i] + bc;
    if (p == 0)      Qs[(row * 8 + r_off) * HS + c_s] = (_Float16)v;
    else if (p == 1) Ks[(row * 8 + r_off) * HS + c_s] = (_Float16)v;
    else             Vt[c_s * N_TOK + row * 8 + r_off] = (_Float16)v;
  }
}

// ---------------------------------------------------------------------------
// Kernel 3: split-K flash attention over N=16384, d=32.
// Grid: x = 128 q-groups (8 waves x 16 q-rows), y = 8 key-groups of 2048 keys.
// K/V chunks of 64 keys double-buffered in LDS via async global->LDS copies,
// shared by the 8 waves of the WG. Each WG emits an UNNORMALIZED partial
// (acc, m, l) for its key range; merge_kernel combines the groups.
// ---------------------------------------------------------------------------
__global__ __launch_bounds__(256) void attn_kernel(
    const _Float16* __restrict__ Qs, const _Float16* __restrict__ Ks,
    const _Float16* __restrict__ Vt,
    float* __restrict__ pacc,    // [1024][KGRP][16][32]
    float* __restrict__ pm,      // [1024][KGRP][16]
    float* __restrict__ pl) {    // [1024][KGRP][16]
  __shared__ __align__(16) _Float16 kbuf[2][64 * HS];   // [chunk key][d]
  __shared__ __align__(16) _Float16 vbuf[2][HS * 64];   // [d][chunk key]
  __shared__ __align__(16) _Float16 pbuf[8][16 * 64];

  int tid  = threadIdx.x;
  int lane = tid & 31, wave = tid >> 5;
  int h = lane >> 4, ln = lane & 15;
  int qb = blockIdx.x * 8 + wave;              // q-block 0..1023
  int q0 = qb * 16;
  int g  = blockIdx.y;                         // key group
  int kbase = g * KEYS_PER_GRP;

  // Q tile in A layout (persistent across the K loop)
  V16 qa;
  const _Float16* qrow = Qs + (q0 + ln) * HS;
  qa.h8[0] = *(const v8h*)(qrow + 16 * h);
  qa.h8[1] = *(const v8h*)(qrow + 16 * h + 8);

  float m[8], l[8];
  v8f acc0 = {}, acc1 = {};
  #pragma unroll
  for (int i = 0; i < 8; i++) { m[i] = -__builtin_inff(); l[i] = 0.0f; }
  const float scale = 0.0625f;                 // 1/sqrt(256)

  _Float16* pb = pbuf[wave];
  const int vc = tid >> 3, vj = tid & 7;       // V-copy assignment per thread

  // prologue: start chunk 0 (2 async ops per wave: 512B each)
  async_b128(&kbuf[0][tid * 8], Ks + kbase * HS + tid * 8);
  async_b128(&vbuf[0][vc * 64 + vj * 8], Vt + vc * N_TOK + kbase + vj * 8);

  const int NCHUNK = KEYS_PER_GRP / 64;        // 32
  for (int it = 0; it < NCHUNK; ++it) {
    int buf = it & 1;
    // prefetch next chunk into the other buffer (safe: it was last read in
    // iteration it-1, which ended with a barrier)
    if (it + 1 < NCHUNK) {
      int nk = kbase + (it + 1) * 64;
      async_b128(&kbuf[buf ^ 1][tid * 8], Ks + nk * HS + tid * 8);
      async_b128(&vbuf[buf ^ 1][vc * 64 + vj * 8], Vt + vc * N_TOK + nk + vj * 8);
      asm volatile("s_wait_asynccnt 2" ::: "memory");  // current chunk landed
    } else {
      asm volatile("s_wait_asynccnt 0" ::: "memory");
    }
    __syncthreads();                                    // publish chunk to WG

    // ---- S = Q @ K^T for 64 keys (4 tiles) ----
    v8f s[4];
    #pragma unroll
    for (int t = 0; t < 4; t++) {
      V16 kb;                                  // B: col ln = key row, K=16h..
      const _Float16* kr = &kbuf[buf][(t * 16 + ln) * HS + 16 * h];
      kb.h8[0] = *(const v8h*)(kr);
      kb.h8[1] = *(const v8h*)(kr + 8);
      v8f z = {};
      s[t] = wmma_f16(qa.v, kb.v, z);
    }
    // ---- online softmax (per row i+8h; stats replicated across 16-lane half)
    #pragma unroll
    for (int i = 0; i < 8; i++) {
      float bm = fmaxf(fmaxf(s[0][i], s[1][i]), fmaxf(s[2][i], s[3][i])) * scale;
      bm = fmaxf(bm, __shfl_xor(bm, 1));
      bm = fmaxf(bm, __shfl_xor(bm, 2));
      bm = fmaxf(bm, __shfl_xor(bm, 4));
      bm = fmaxf(bm, __shfl_xor(bm, 8));
      float mn    = fmaxf(m[i], bm);
      float alpha = __expf(m[i] - mn);
      float rs = 0.0f;
      #pragma unroll
      for (int t = 0; t < 4; t++) {
        float pv = __expf(s[t][i] * scale - mn);
        s[t][i] = pv;
        rs += pv;
      }
      rs += __shfl_xor(rs, 1); rs += __shfl_xor(rs, 2);
      rs += __shfl_xor(rs, 4); rs += __shfl_xor(rs, 8);
      l[i] = l[i] * alpha + rs;
      m[i] = mn;
      acc0[i] = acc0[i] * alpha;
      acc1[i] = acc1[i] * alpha;
    }
    // ---- P: C layout f32 -> LDS -> A layout f16 (private per wave) ----
    #pragma unroll
    for (int t = 0; t < 4; t++)
      #pragma unroll
      for (int i = 0; i < 8; i++)
        pb[(i + 8 * h) * 64 + t * 16 + ln] = (_Float16)s[t][i];
    asm volatile("s_wait_dscnt 0" ::: "memory");

    V16 pa0, pa1;
    const _Float16* prow = pb + ln * 64;
    pa0.h8[0] = *(const v8h*)(prow + 16 * h);
    pa0.h8[1] = *(const v8h*)(prow + 16 * h + 8);
    pa1.h8[0] = *(const v8h*)(prow + 32 + 16 * h);
    pa1.h8[1] = *(const v8h*)(prow + 40 + 16 * h);

    // ---- acc += P @ V (two K-chunks of 32, two N-tiles of 16) ----
    #pragma unroll
    for (int kk = 0; kk < 2; kk++) {
      V16 vb0, vb1;                            // B: col of V = vbuf row, contiguous
      const _Float16* v0 = &vbuf[buf][(ln)      * 64 + kk * 32 + 16 * h];
      const _Float16* v1 = &vbuf[buf][(16 + ln) * 64 + kk * 32 + 16 * h];
      vb0.h8[0] = *(const v8h*)(v0); vb0.h8[1] = *(const v8h*)(v0 + 8);
      vb1.h8[0] = *(const v8h*)(v1); vb1.h8[1] = *(const v8h*)(v1 + 8);
      v16h pav = kk ? pa1.v : pa0.v;
      acc0 = wmma_f16(pav, vb0.v, acc0);
      acc1 = wmma_f16(pav, vb1.v, acc1);
    }
    __syncthreads();       // everyone done reading buf before it is refilled
  }

  // ---- epilogue: emit unnormalized partial for this key group ----
  float* pa = pacc + (size_t)(qb * KGRP + g) * (16 * 32);
  #pragma unroll
  for (int i = 0; i < 8; i++) {
    int row = i + 8 * h;                       // row within q-block
    pa[row * 32 + ln]      = acc0[i];
    pa[row * 32 + 16 + ln] = acc1[i];
    if (ln == 0) {
      pm[(qb * KGRP + g) * 16 + row] = m[i];
      pl[(qb * KGRP + g) * 16 + row] = l[i];
    }
  }
}

// ---------------------------------------------------------------------------
// Kernel 3b: merge the KGRP split-K partials (log-sum-exp combine), then
// concat-heads scatter to attended f32 output + f16 copy for the final GEMM.
// One thread per (attended row, col): 16384*32 threads.
// ---------------------------------------------------------------------------
__global__ __launch_bounds__(256) void merge_kernel(
    const float* __restrict__ pacc, const float* __restrict__ pm,
    const float* __restrict__ pl,
    float* __restrict__ att_out, _Float16* __restrict__ att_h) {
  int idx = blockIdx.x * 256 + threadIdx.x;    // 0 .. 524287
  int r = idx >> 5, c = idx & 31;              // attended row, col
  int qb = r >> 4, ri = r & 15;
  int base = qb * KGRP;

  float mstar = -__builtin_inff();
  #pragma unroll
  for (int g = 0; g < KGRP; g++)
    mstar = fmaxf(mstar, pm[(base + g) * 16 + ri]);
  float num = 0.0f, den = 0.0f;
  #pragma unroll
  for (int g = 0; g < KGRP; g++) {
    float sc = __expf(pm[(base + g) * 16 + ri] - mstar);
    den += sc * pl[(base + g) * 16 + ri];
    num += sc * pacc[(size_t)(base + g) * (16 * 32) + ri * 32 + c];
  }
  float a = num / den;
  int b = r >> 3, hh = r & 7;                  // out[b, c*8+hh] = att[r, c]
  att_out[b * F + c * 8 + hh] = a;
  att_h[b * F + c * 8 + hh]   = (_Float16)a;
}

// ---------------------------------------------------------------------------
// Kernel 4: outputs = attended @ Wo + bo   (WMMA, one wave per 16x16 tile)
// ---------------------------------------------------------------------------
__global__ __launch_bounds__(256) void outproj_kernel(
    const _Float16* __restrict__ ah, const _Float16* __restrict__ Wot,
    const float* __restrict__ bo, float* __restrict__ out) {
  int lane = threadIdx.x & 31, wave = threadIdx.x >> 5;
  int wid = blockIdx.x * 8 + wave;             // 0 .. 2047
  int m0 = (wid >> 4) * 16, n0 = (wid & 15) * 16;
  int h = lane >> 4, ln = lane & 15;

  v8f c = {};
  #pragma unroll
  for (int k0 = 0; k0 < F; k0 += 32) {
    V16 a, b;
    const _Float16* ar = ah + (m0 + ln) * F + k0;
    a.h8[0] = *(const v8h*)(ar + 16 * h);
    a.h8[1] = *(const v8h*)(ar + 16 * h + 8);
    const _Float16* br = Wot + (n0 + ln) * F + k0 + 16 * h;
    b.h8[0] = *(const v8h*)(br);
    b.h8[1] = *(const v8h*)(br + 8);
    c = wmma_f16(a.v, b.v, c);
  }
  float bc = bo[n0 + ln];
  #pragma unroll
  for (int i = 0; i < 8; i++) {
    int row = m0 + i + 8 * h;
    out[row * F + n0 + ln] = c[i] + bc;
  }
}

// ---------------------------------------------------------------------------
extern "C" void kernel_launch(void* const* d_in, const int* in_sizes, int n_in,
                              void* d_out, int out_size, void* d_ws, size_t ws_size,
                              hipStream_t stream) {
  (void)in_sizes; (void)n_in; (void)out_size; (void)ws_size;
  const float* x  = (const float*)d_in[0];
  const float* Wq = (const float*)d_in[1];
  const float* bq = (const float*)d_in[2];
  const float* Wk = (const float*)d_in[3];
  const float* bk = (const float*)d_in[4];
  const float* Wv = (const float*)d_in[5];
  const float* bv = (const float*)d_in[6];
  const float* Wo = (const float*)d_in[7];
  const float* bo = (const float*)d_in[8];
  float* out      = (float*)d_out;                 // [2048,256] outputs
  float* att_out  = out + BATCH * F;               // [2048,256] attended

  _Float16* ws = (_Float16*)d_ws;
  _Float16* xh  = ws;                              // 524288 f16
  _Float16* Wt  = ws + 524288;                     // 262144 (4 transposed W)
  _Float16* Qs  = ws + 786432;                     // 524288
  _Float16* Ks  = ws + 1310720;                    // 524288
  _Float16* Vt  = ws + 1835008;                    // 524288 (transposed [32,16384])
  _Float16* ath = ws + 2359296;                    // 524288
  float* wsf    = (float*)(ws + 2883584);          // f32 region (16B aligned)
  float* pacc   = wsf;                             // 1024*8*16*32 = 4194304 f32
  float* pm     = wsf + 4194304;                   // 131072 f32
  float* pl     = pm + 131072;                     // 131072 f32

  prep_kernel<<<2048, 256, 0, stream>>>(x, Wq, Wk, Wv, Wo, xh, Wt);
  proj_kernel<<<768, 256, 0, stream>>>(xh, Wt, bq, bk, bv, Qs, Ks, Vt);
  attn_kernel<<<dim3(128, KGRP), 256, 0, stream>>>(Qs, Ks, Vt, pacc, pm, pl);
  merge_kernel<<<2048, 256, 0, stream>>>(pacc, pm, pl, att_out, ath);
  outproj_kernel<<<256, 256, 0, stream>>>(ath, Wt + 3 * F * F, bo, out);
}
